// MultiHeadAttention_87162066305355
// MI455X (gfx1250) — matrix operand order repaired and checked by reading, not compiled, not run on hardware
//
#include <hip/hip_runtime.h>

// ---------------------------------------------------------------------------
// MHA forward for MI455X (gfx1250, wave32, WMMA).
// bf16x3 split GEMMs (hi/lo error compensation, fp32 accumulate) on
// v_wmma_f32_16x16x32_bf16; flash-attention streaming softmax.
// Operand fragments: per lane two 16B row runs (k=8h+i, 16+8h+i) in LDS -> ds_load_b128.
// Staging: CDNA5 async-to-LDS (global_load_async_to_lds_b128, ASYNCcnt),
// now double-buffered so asyncs for tile i+1 overlap the WMMA burst on tile i.
// ---------------------------------------------------------------------------

#define D_MODEL 2048
#define N_HEAD  16
#define D_HEAD  128
#define BATCH   2
#define SEQ     2048
#define M_TOK   (BATCH * SEQ)   // 4096

typedef __attribute__((ext_vector_type(16))) __bf16        v16bf;
typedef __attribute__((ext_vector_type(8)))  float         v8f;
typedef __attribute__((ext_vector_type(4)))  unsigned int  v4u;

union ABReg {
    v16bf v;
    unsigned short u[16];
    v4u q[2];
};

static __device__ inline unsigned short f2bf(float f) {
    unsigned u = __float_as_uint(f);
    u += 0x7FFFu + ((u >> 16) & 1u);      // round-to-nearest-even
    return (unsigned short)(u >> 16);
}
static __device__ inline float bf2f(unsigned short h) {
    return __uint_as_float(((unsigned)h) << 16);
}
static __device__ inline v8f v8f_zero() {
    v8f z;
    for (int i = 0; i < 8; ++i) z[i] = 0.0f;
    return z;
}
static __device__ inline v8f wmma_bf16(const ABReg& a, const ABReg& b, v8f c) {
    return __builtin_amdgcn_wmma_f32_16x16x32_bf16(
        false, a.v, false, b.v, (short)0, c, false, false);
}

// CDNA5 async global->LDS copy (GLOBAL_LOAD_ASYNC_TO_LDS_B128, GV mode).
static __device__ inline void async_ld_b128(unsigned short* lptr,
                                            const unsigned short* gptr) {
    unsigned loff = (unsigned)(unsigned long long)lptr;
    asm volatile("global_load_async_to_lds_b128 %0, %1, off"
                 :: "v"(loff), "v"(gptr)
                 : "memory");
}
static __device__ inline void wait_async0() {
    asm volatile("s_wait_asynccnt 0" ::: "memory");
}

// ---------------------------------------------------------------------------
// fp32 -> (bf16 hi, bf16 lo) elementwise split
// ---------------------------------------------------------------------------
__global__ void split_bf16_kernel(const float* __restrict__ src,
                                  unsigned short* __restrict__ hi,
                                  unsigned short* __restrict__ lo, int n) {
    int i = blockIdx.x * blockDim.x + threadIdx.x;
    int stride = gridDim.x * blockDim.x;
    for (; i < n; i += stride) {
        float v = src[i];
        unsigned short h = f2bf(v);
        hi[i] = h;
        lo[i] = f2bf(v - bf2f(h));
    }
}

// ---------------------------------------------------------------------------
// fp32 W[k][n] -> split-bf16 WT[n][k]  (tiled transpose, coalesced both ways)
// ---------------------------------------------------------------------------
__global__ __launch_bounds__(256)
void split_transpose_kernel(const float* __restrict__ W,
                            unsigned short* __restrict__ Th,
                            unsigned short* __restrict__ Tl) {
    __shared__ float tile[32][33];
    const int tx = threadIdx.x & 31;
    const int ty = threadIdx.x >> 5;     // 0..7
    const int k0 = blockIdx.x * 32;
    const int n0 = blockIdx.y * 32;
#pragma unroll
    for (int i = 0; i < 4; ++i) {
        int r = ty * 4 + i;
        tile[r][tx] = W[(size_t)(k0 + r) * D_MODEL + n0 + tx];
    }
    __syncthreads();
#pragma unroll
    for (int i = 0; i < 4; ++i) {
        int r = ty * 4 + i;                 // n offset within tile
        float v = tile[tx][r];              // = W[k0+tx][n0+r]
        size_t o = (size_t)(n0 + r) * D_MODEL + k0 + tx;
        unsigned short hv = f2bf(v);
        Th[o] = hv;
        Tl[o] = f2bf(v - bf2f(hv));
    }
}

// ---------------------------------------------------------------------------
// GEMM: O = A[M,2048] * W[2048,2048] + bias, W given transposed (WT[n][k]).
// 256 threads = 8 waves, wave tile 32Mx64N, WG tile 64Mx256N, bf16x3.
// Double-buffered async staging: tile kk+32 streams in during compute on kk.
// mode 0: split-bf16 out [B,H,S,d];  mode 1: split-bf16 out [B,H,d,S];
// mode 2: fp32 out row-major [M, D_MODEL].
// ---------------------------------------------------------------------------
__global__ __launch_bounds__(256)
void proj_gemm_kernel(const unsigned short* __restrict__ Ah,
                      const unsigned short* __restrict__ Al,
                      const unsigned short* __restrict__ WTh,
                      const unsigned short* __restrict__ WTl,
                      const float* __restrict__ bias,
                      unsigned short* __restrict__ Oh,
                      unsigned short* __restrict__ Ol,
                      float* __restrict__ Of,
                      int mode) {
    __shared__ __align__(16) unsigned short xs_h[2][64][40],  xs_l[2][64][40];
    __shared__ __align__(16) unsigned short ws_h[2][256][40], ws_l[2][256][40];

    const int tid  = threadIdx.x;
    const int lane = tid & 31;
    const int wave = tid >> 5;
    const int wm   = wave & 1;
    const int wn   = wave >> 1;
    const int hh   = lane >> 4;
    const int l16  = lane & 15;
    const int nbase = blockIdx.x * 256;
    const int mbase = blockIdx.y * 64;

    auto stage = [&](int buf, int kkp) {
        {
            int row = tid >> 2, qc = tid & 3;
            size_t g = (size_t)(mbase + row) * D_MODEL + kkp + qc * 8;
            async_ld_b128(&xs_h[buf][row][qc * 8], Ah + g);
            async_ld_b128(&xs_l[buf][row][qc * 8], Al + g);
        }
#pragma unroll
        for (int t = 0; t < 4; ++t) {
            int idx = tid + t * 256;
            int row = idx >> 2, qc = idx & 3;
            size_t g = (size_t)(nbase + row) * D_MODEL + kkp + qc * 8;
            async_ld_b128(&ws_h[buf][row][qc * 8], WTh + g);
            async_ld_b128(&ws_l[buf][row][qc * 8], WTl + g);
        }
    };

    v8f acc[2][4];
#pragma unroll
    for (int m2 = 0; m2 < 2; ++m2)
#pragma unroll
        for (int t = 0; t < 4; ++t) acc[m2][t] = v8f_zero();

    stage(0, 0);
    wait_async0();
    __syncthreads();

    for (int kk = 0; kk < D_MODEL; kk += 32) {
        const int buf = (kk >> 5) & 1;
        if (kk + 32 < D_MODEL)
            stage(buf ^ 1, kk + 32);       // in flight during the WMMA burst

        // ---- A fragments: u[0..7] K=8h+i, u[8..15] K=16+8h+i ----
        ABReg ah[2], al[2];
#pragma unroll
        for (int m2 = 0; m2 < 2; ++m2) {
            const v4u* ph = (const v4u*)&xs_h[buf][wm * 32 + m2 * 16 + l16][0];
            const v4u* pl = (const v4u*)&xs_l[buf][wm * 32 + m2 * 16 + l16][0];
            ah[m2].q[0] = ph[hh];     ah[m2].q[1] = ph[2 + hh];
            al[m2].q[0] = pl[hh];     al[m2].q[1] = pl[2 + hh];
        }
        // ---- B fragments: u[0..7] K=8h+i, u[8..15] K=16+8h+i ----
#pragma unroll
        for (int t = 0; t < 4; ++t) {
            ABReg bh, bl;
            const v4u* pbh = (const v4u*)&ws_h[buf][wn * 64 + t * 16 + l16][0];
            const v4u* pbl = (const v4u*)&ws_l[buf][wn * 64 + t * 16 + l16][0];
            bh.q[0] = pbh[hh];    bh.q[1] = pbh[2 + hh];
            bl.q[0] = pbl[hh];    bl.q[1] = pbl[2 + hh];
#pragma unroll
            for (int m2 = 0; m2 < 2; ++m2) {
                acc[m2][t] = wmma_bf16(ah[m2], bh, acc[m2][t]);
                acc[m2][t] = wmma_bf16(ah[m2], bl, acc[m2][t]);
                acc[m2][t] = wmma_bf16(al[m2], bh, acc[m2][t]);
            }
        }

        wait_async0();     // next tile resident
        __syncthreads();   // all waves done reading buf / writing buf^1
    }

    // ---- epilogue: bias + store ----
#pragma unroll
    for (int m2 = 0; m2 < 2; ++m2)
#pragma unroll
        for (int t = 0; t < 4; ++t) {
            const int ncol = nbase + wn * 64 + t * 16 + l16;
            const float bv = bias[ncol];
#pragma unroll
            for (int j = 0; j < 8; ++j) {
                const int mrow = mbase + wm * 32 + m2 * 16 + j + 8 * hh;
                float v = acc[m2][t][j] + bv;
                if (mode == 2) {
                    Of[(size_t)mrow * D_MODEL + ncol] = v;
                } else {
                    int b = mrow >> 11, s = mrow & (SEQ - 1);
                    int h = ncol >> 7,  d = ncol & (D_HEAD - 1);
                    size_t o = (mode == 0)
                        ? (((size_t)(b * N_HEAD + h) * SEQ) + s) * D_HEAD + d
                        : (((size_t)(b * N_HEAD + h) * D_HEAD) + d) * SEQ + s;
                    unsigned short hv = f2bf(v);
                    Oh[o] = hv;
                    Ol[o] = f2bf(v - bf2f(hv));
                }
            }
        }
}

// ---------------------------------------------------------------------------
// Flash attention: 128 threads = 4 waves, wave owns 16 q-rows of one (b,h).
// Double-buffered async K/V^T staging overlapping the 48-WMMA burst.
// ---------------------------------------------------------------------------
__global__ __launch_bounds__(128)
void attn_kernel(const unsigned short* __restrict__ Qh,  const unsigned short* __restrict__ Ql,
                 const unsigned short* __restrict__ Kh,  const unsigned short* __restrict__ Kl,
                 const unsigned short* __restrict__ Vth, const unsigned short* __restrict__ Vtl,
                 const float* __restrict__ mask,
                 unsigned short* __restrict__ Ch, unsigned short* __restrict__ Cl) {
    __shared__ __align__(16) unsigned short kh_s[2][32][136], kl_s[2][32][136];
    __shared__ __align__(16) unsigned short vt_h[2][128][40], vt_l[2][128][40];
    __shared__ __align__(16) unsigned short ph_s[4][16][40],  pl_s[4][16][40];

    const int tid  = threadIdx.x;
    const int lane = tid & 31;
    const int wave = tid >> 5;
    const int hh   = lane >> 4;
    const int l16  = lane & 15;
    const int bh   = blockIdx.y;
    const int qbase = blockIdx.x * 64 + wave * 16;
    const size_t base = (size_t)bh * SEQ * D_HEAD;
    const float scale = 0.08838834764831845f;      // 1/sqrt(128)

    auto stage_kv = [&](int buf, int ktp) {
#pragma unroll
        for (int t = 0; t < 4; ++t) {
            int idx = tid + t * 128;
            {   int r = idx >> 4, qc = idx & 15;
                size_t g = base + (size_t)(ktp + r) * D_HEAD + qc * 8;
                async_ld_b128(&kh_s[buf][r][qc * 8], Kh + g);
                async_ld_b128(&kl_s[buf][r][qc * 8], Kl + g);
            }
            {   int r = idx >> 2, qc = idx & 3;
                size_t g = base + (size_t)r * SEQ + ktp + qc * 8;
                async_ld_b128(&vt_h[buf][r][qc * 8], Vth + g);
                async_ld_b128(&vt_l[buf][r][qc * 8], Vtl + g);
            }
        }
    };

    // Q held in registers: 4 chunks (16x32 over d), hi+lo
    ABReg qa_h[4], qa_l[4];
    {
        const int qrow = qbase + l16;
        const v4u* gh = (const v4u*)(Qh + base + (size_t)qrow * D_HEAD);
        const v4u* gl = (const v4u*)(Ql + base + (size_t)qrow * D_HEAD);
#pragma unroll
        for (int c = 0; c < 4; ++c) {
            qa_h[c].q[0] = gh[4 * c + hh];  qa_h[c].q[1] = gh[4 * c + 2 + hh];
            qa_l[c].q[0] = gl[4 * c + hh];  qa_l[c].q[1] = gl[4 * c + 2 + hh];
        }
    }

    v8f acc[8];
#pragma unroll
    for (int t = 0; t < 8; ++t) acc[t] = v8f_zero();
    float mrun[8], lrun[8];
#pragma unroll
    for (int j = 0; j < 8; ++j) { mrun[j] = -3.0e38f; lrun[j] = 0.0f; }

    stage_kv(0, 0);
    wait_async0();
    __syncthreads();

    for (int kt = 0; kt < SEQ; kt += 32) {
        const int buf = (kt >> 5) & 1;
        if (kt + 32 < SEQ)
            stage_kv(buf ^ 1, kt + 32);    // in flight during compute

        // ---- scores S = Q K^T : two 16x16 C tiles, bf16x3 ----
        v8f s0 = v8f_zero(), s1 = v8f_zero();
#pragma unroll
        for (int c = 0; c < 4; ++c) {
            ABReg bh0, bl0, bh1, bl1;
            const v4u* p0h = (const v4u*)&kh_s[buf][l16][0];
            const v4u* p0l = (const v4u*)&kl_s[buf][l16][0];
            const v4u* p1h = (const v4u*)&kh_s[buf][16 + l16][0];
            const v4u* p1l = (const v4u*)&kl_s[buf][16 + l16][0];
            const int qi = 4 * c + hh;
            bh0.q[0] = p0h[qi]; bh0.q[1] = p0h[qi + 2];
            bl0.q[0] = p0l[qi]; bl0.q[1] = p0l[qi + 2];
            bh1.q[0] = p1h[qi]; bh1.q[1] = p1h[qi + 2];
            bl1.q[0] = p1l[qi]; bl1.q[1] = p1l[qi + 2];
            s0 = wmma_bf16(qa_h[c], bh0, s0);
            s0 = wmma_bf16(qa_h[c], bl0, s0);
            s0 = wmma_bf16(qa_l[c], bh0, s0);
            s1 = wmma_bf16(qa_h[c], bh1, s1);
            s1 = wmma_bf16(qa_h[c], bl1, s1);
            s1 = wmma_bf16(qa_l[c], bh1, s1);
        }

        // ---- scale + additive mask ----
#pragma unroll
        for (int j = 0; j < 8; ++j) {
            const int qr = qbase + j + 8 * hh;
            const float* mrow = mask + (size_t)qr * SEQ + kt;
            s0[j] = s0[j] * scale + mrow[l16];
            s1[j] = s1[j] * scale + mrow[16 + l16];
        }

        // ---- online softmax (row reductions: width-16 shuffles) ----
        float corr[8];
#pragma unroll
        for (int j = 0; j < 8; ++j) {
            float m = fmaxf(s0[j], s1[j]);
#pragma unroll
            for (int off = 8; off >= 1; off >>= 1)
                m = fmaxf(m, __shfl_xor(m, off, 16));
            float mnew = fmaxf(mrun[j], m);
            corr[j] = __expf(mrun[j] - mnew);
            float p0 = __expf(s0[j] - mnew);
            float p1 = __expf(s1[j] - mnew);
            s0[j] = p0; s1[j] = p1;
            float rs = p0 + p1;
#pragma unroll
            for (int off = 8; off >= 1; off >>= 1)
                rs += __shfl_xor(rs, off, 16);
            lrun[j] = lrun[j] * corr[j] + rs;
            mrun[j] = mnew;
        }
#pragma unroll
        for (int t = 0; t < 8; ++t)
#pragma unroll
            for (int j = 0; j < 8; ++j)
                acc[t][j] = acc[t][j] * corr[j];

        // ---- P (C layout) -> per-wave LDS -> A layout, split hi/lo ----
#pragma unroll
        for (int j = 0; j < 8; ++j) {
            const int r = j + 8 * hh;
            unsigned short h0 = f2bf(s0[j]);
            ph_s[wave][r][l16] = h0;
            pl_s[wave][r][l16] = f2bf(s0[j] - bf2f(h0));
            unsigned short h1 = f2bf(s1[j]);
            ph_s[wave][r][16 + l16] = h1;
            pl_s[wave][r][16 + l16] = f2bf(s1[j] - bf2f(h1));
        }
        asm volatile("s_wait_dscnt 0" ::: "memory");   // wave-local LDS RAW

        ABReg pa_h, pa_l;
        {
            const v4u* pp_h = (const v4u*)&ph_s[wave][l16][0];
            const v4u* pp_l = (const v4u*)&pl_s[wave][l16][0];
            pa_h.q[0] = pp_h[hh];  pa_h.q[1] = pp_h[2 + hh];
            pa_l.q[0] = pp_l[hh];  pa_l.q[1] = pp_l[2 + hh];
        }

        // ---- ctx += P V : 8 d-tiles, B from V^T rows, bf16x3 ----
#pragma unroll
        for (int t = 0; t < 8; ++t) {
            ABReg bvh, bvl;
            const v4u* pvh = (const v4u*)&vt_h[buf][t * 16 + l16][0];
            const v4u* pvl = (const v4u*)&vt_l[buf][t * 16 + l16][0];
            bvh.q[0] = pvh[hh];  bvh.q[1] = pvh[2 + hh];
            bvl.q[0] = pvl[hh];  bvl.q[1] = pvl[2 + hh];
            acc[t] = wmma_bf16(pa_h, bvh, acc[t]);
            acc[t] = wmma_bf16(pa_l, bvh, acc[t]);
            acc[t] = wmma_bf16(pa_h, bvl, acc[t]);
        }

        wait_async0();     // next K/V tile resident
        __syncthreads();
    }

    // ---- finalize: /= l, split-bf16 ctx in [M_TOK, D_MODEL] ----
    const int b = bh / N_HEAD, h = bh % N_HEAD;
#pragma unroll
    for (int j = 0; j < 8; ++j) {
        const int qr = qbase + j + 8 * hh;
        const float inv = 1.0f / lrun[j];
        const int mtok = b * SEQ + qr;
#pragma unroll
        for (int t = 0; t < 8; ++t) {
            float v = acc[t][j] * inv;
            size_t o = (size_t)mtok * D_MODEL + h * D_HEAD + t * 16 + l16;
            unsigned short hv = f2bf(v);
            Ch[o] = hv;
            Cl[o] = f2bf(v - bf2f(hv));
        }
    }
}

// ---------------------------------------------------------------------------
extern "C" void kernel_launch(void* const* d_in, const int* in_sizes, int n_in,
                              void* d_out, int out_size, void* d_ws, size_t ws_size,
                              hipStream_t stream) {
    (void)in_sizes; (void)n_in; (void)out_size; (void)ws_size;
    const float* x    = (const float*)d_in[0];
    const float* mask = (const float*)d_in[1];
    const float* wq = (const float*)d_in[2];  const float* bq = (const float*)d_in[3];
    const float* wk = (const float*)d_in[4];  const float* bk = (const float*)d_in[5];
    const float* wv = (const float*)d_in[6];  const float* bv = (const float*)d_in[7];
    const float* wo = (const float*)d_in[8];  const float* bo = (const float*)d_in[9];
    float* out = (float*)d_out;

    const size_t nx = (size_t)M_TOK * D_MODEL;
    const size_t nw = (size_t)D_MODEL * D_MODEL;
    const size_t nq = (size_t)BATCH * N_HEAD * SEQ * D_HEAD;

    unsigned short* p   = (unsigned short*)d_ws;
    unsigned short* xh  = p; p += nx;   unsigned short* xl  = p; p += nx;
    unsigned short* wqh = p; p += nw;   unsigned short* wql = p; p += nw;  // transposed
    unsigned short* wkh = p; p += nw;   unsigned short* wkl = p; p += nw;  // transposed
    unsigned short* wvh = p; p += nw;   unsigned short* wvl = p; p += nw;  // transposed
    unsigned short* woh = p; p += nw;   unsigned short* wol = p; p += nw;  // transposed
    unsigned short* Qh  = p; p += nq;   unsigned short* Ql  = p; p += nq;
    unsigned short* Kh  = p; p += nq;   unsigned short* Kl  = p; p += nq;
    unsigned short* Vth = p; p += nq;   unsigned short* Vtl = p; p += nq;  // [B,H,d,S]
    unsigned short* Ch  = p; p += nx;   unsigned short* Cl  = p; p += nx;

    // 1) conversions
    split_bf16_kernel<<<2048, 256, 0, stream>>>(x, xh, xl, (int)nx);
    dim3 tg(D_MODEL / 32, D_MODEL / 32);
    split_transpose_kernel<<<tg, 256, 0, stream>>>(wq, wqh, wql);
    split_transpose_kernel<<<tg, 256, 0, stream>>>(wk, wkh, wkl);
    split_transpose_kernel<<<tg, 256, 0, stream>>>(wv, wvh, wvl);
    split_transpose_kernel<<<tg, 256, 0, stream>>>(wo, woh, wol);

    // 2) projections
    dim3 pg(D_MODEL / 256, M_TOK / 64);
    proj_gemm_kernel<<<pg, 256, 0, stream>>>(xh, xl, wqh, wql, bq, Qh, Ql, nullptr, 0);
    proj_gemm_kernel<<<pg, 256, 0, stream>>>(xh, xl, wkh, wkl, bk, Kh, Kl, nullptr, 0);
    proj_gemm_kernel<<<pg, 256, 0, stream>>>(xh, xl, wvh, wvl, bv, Vth, Vtl, nullptr, 1);

    // 3) flash attention -> split-bf16 ctx [M_TOK, D_MODEL]
    dim3 ag(SEQ / 64, BATCH * N_HEAD);
    attn_kernel<<<ag, 128, 0, stream>>>(Qh, Ql, Kh, Kl, Vth, Vtl, mask, Ch, Cl);

    // 4) output projection -> fp32 d_out
    proj_gemm_kernel<<<pg, 256, 0, stream>>>(Ch, Cl, woh, wol, bo, nullptr, nullptr, out, 2);
}
